// NetVLAD_25116968747409
// MI455X (gfx1250) — matrix operand order repaired and checked by reading, not compiled, not run on hardware
//
#include <hip/hip_runtime.h>

typedef _Float16 v16h __attribute__((ext_vector_type(16)));
typedef _Float16 h8   __attribute__((ext_vector_type(8)));
typedef float    v8f  __attribute__((ext_vector_type(8)));

#define N_IMG   64
#define C_DIM   128
#define K_CL    64
#define S_PIX   3600
#define T_PIX   32
#define NT      113      // ceil(3600/32)
#define CHUNKS  8
#define CW_PAD  136      // 272B row stride: 16B aligned, bank-spread
#define XNT_PAD 136
#define XN_PAD  40       // 80B row stride
#define A_PAD   40

__device__ __forceinline__ v16h cat16(h8 lo, h8 hi) {
  return __builtin_shufflevector(lo, hi, 0,1,2,3,4,5,6,7,8,9,10,11,12,13,14,15);
}

__global__ __launch_bounds__(256)
void netvlad_zero(float* __restrict__ ws, int n) {
  int idx = blockIdx.x * 256 + threadIdx.x;
  for (int i = idx; i < n; i += gridDim.x * 256) ws[i] = 0.0f;
}

__global__ __launch_bounds__(256)
void netvlad_main(const float* __restrict__ x, const float* __restrict__ conv_w,
                  float* __restrict__ vlad_ws, float* __restrict__ asum_ws) {
  __shared__ alignas(16) _Float16 cwL [K_CL][CW_PAD];    // conv_w f16, [k][c]
  __shared__ alignas(16) _Float16 xnL [C_DIM][XN_PAD];   // xn f16, [c][t]
  __shared__ alignas(16) _Float16 xnTL[T_PIX][XNT_PAD];  // xn f16, [t][c]
  __shared__ alignas(16) _Float16 aL  [K_CL][A_PAD];     // soft-assign f16, [k][t]
  __shared__ alignas(16) float    lgL [K_CL][T_PIX];     // logits f32 (reused for asum)
  __shared__ float redA[8][T_PIX];
  __shared__ float redB[8][T_PIX];

  const int tid   = threadIdx.x;
  const int n     = blockIdx.x / CHUNKS;
  const int chunk = blockIdx.x % CHUNKS;
  const int w    = tid >> 5;          // wave id 0..7
  const int lane = tid & 31;
  const int g    = lane >> 4;         // half-wave group
  const int l    = lane & 15;
  const int p    = lane;              // pixel-in-tile for phases 1/3
  const int grp  = w;                 // 8 groups of 8 k's for softmax

  // stage conv_w into LDS as f16
  for (int i = tid; i < K_CL * C_DIM; i += 256) {
    cwL[i >> 7][i & 127] = (_Float16)conv_w[i];
  }

  const int kt1 = w & 3, tt1 = w >> 2;   // GEMM1: logits tile (k-tile, t-tile)
  const int kt2 = w >> 1, ch2 = w & 1;   // GEMM2: k-tile, c-half (4 c-tiles each)

  v8f   vacc[4] = {};                    // persistent vlad accumulators (4 tiles)
  float asumR[8] = {};                   // per-thread partial sum of a over pixels

  const float* xb = x + (size_t)n * C_DIM * S_PIX;
  __syncthreads();

  for (int tt = chunk; tt < NT; tt += CHUNKS) {
    const int s0 = tt * T_PIX;
    const bool valid = (s0 + p) < S_PIX;

    // ---- phase 1: load x tile, L2 normalize over C, emit f16 in both layouts ----
    float vals[16];
    float ss = 0.0f;
    #pragma unroll
    for (int i = 0; i < 16; ++i) {
      const int c = i * 8 + grp;
      float v = valid ? xb[(size_t)c * S_PIX + s0 + p] : 0.0f;
      vals[i] = v;
      ss += v * v;
    }
    redA[grp][p] = ss;
    __syncthreads();
    float tot = 0.0f;
    #pragma unroll
    for (int j = 0; j < 8; ++j) tot += redA[j][p];
    const float inv = 1.0f / fmaxf(sqrtf(tot), 1e-12f);
    #pragma unroll
    for (int i = 0; i < 16; ++i) {
      const int c = i * 8 + grp;
      const _Float16 hv = (_Float16)(vals[i] * inv);
      xnL[c][p]  = hv;
      xnTL[p][c] = hv;
    }
    __syncthreads();

    // ---- phase 2: GEMM1  logits[64x32] = W[64x128] @ xn[128x32] ----
    v8f lac = {};
    #pragma unroll
    for (int step = 0; step < 4; ++step) {
      const int cb = step * 32;
      h8 alo = *(const h8*)&cwL[kt1 * 16 + l][cb + g * 8];
      h8 ahi = *(const h8*)&cwL[kt1 * 16 + l][cb + 16 + g * 8];
      h8 blo = *(const h8*)&xnTL[tt1 * 16 + l][cb + g * 8];
      h8 bhi = *(const h8*)&xnTL[tt1 * 16 + l][cb + 16 + g * 8];
      lac = __builtin_amdgcn_wmma_f32_16x16x32_f16(
          false, cat16(alo, ahi), false, cat16(blo, bhi),
          (short)0, lac, false, false);
    }
    #pragma unroll
    for (int r = 0; r < 8; ++r)
      lgL[kt1 * 16 + r + g * 8][tt1 * 16 + l] = lac[r];
    __syncthreads();

    // ---- phase 3: softmax over K per pixel (8 threads x 8 k's each) ----
    float lg[8];
    #pragma unroll
    for (int j = 0; j < 8; ++j) lg[j] = lgL[grp * 8 + j][p];
    float pm = lg[0];
    #pragma unroll
    for (int j = 1; j < 8; ++j) pm = fmaxf(pm, lg[j]);
    redB[grp][p] = pm;
    __syncthreads();
    float mm = redB[0][p];
    #pragma unroll
    for (int j = 1; j < 8; ++j) mm = fmaxf(mm, redB[j][p]);
    float ex[8];
    float ps = 0.0f;
    #pragma unroll
    for (int j = 0; j < 8; ++j) { ex[j] = __expf(lg[j] - mm); ps += ex[j]; }
    redA[grp][p] = ps;
    __syncthreads();
    float tsum = 0.0f;
    #pragma unroll
    for (int j = 0; j < 8; ++j) tsum += redA[j][p];
    const float fac = valid ? (1.0f / tsum) : 0.0f;
    #pragma unroll
    for (int j = 0; j < 8; ++j) {
      const float av = ex[j] * fac;
      aL[grp * 8 + j][p] = (_Float16)av;
      asumR[j] += av;
    }
    __syncthreads();

    // ---- phase 4: GEMM2  vlad[64x128] += a[64x32] @ xn^T[32x128] ----
    {
      h8 alo = *(const h8*)&aL[kt2 * 16 + l][g * 8];
      h8 ahi = *(const h8*)&aL[kt2 * 16 + l][16 + g * 8];
      const v16h Am = cat16(alo, ahi);
      #pragma unroll
      for (int j = 0; j < 4; ++j) {
        const int c0 = (ch2 * 4 + j) * 16;
        h8 blo = *(const h8*)&xnL[c0 + l][g * 8];
        h8 bhi = *(const h8*)&xnL[c0 + l][16 + g * 8];
        vacc[j] = __builtin_amdgcn_wmma_f32_16x16x32_f16(
            false, Am, false, cat16(blo, bhi),
            (short)0, vacc[j], false, false);
      }
    }
    __syncthreads();
  }

  // ---- flush vlad accumulators (one atomic per element per chunk) ----
  float* vb = vlad_ws + (size_t)n * K_CL * C_DIM;
  #pragma unroll
  for (int j = 0; j < 4; ++j) {
    const int c0 = (ch2 * 4 + j) * 16 + l;
    #pragma unroll
    for (int r = 0; r < 8; ++r) {
      const int k = kt2 * 16 + r + g * 8;
      atomicAdd(&vb[k * C_DIM + c0], vacc[j][r]);
    }
  }

  // ---- reduce per-k assignment sums across pixels, flush ----
  #pragma unroll
  for (int j = 0; j < 8; ++j) lgL[grp * 8 + j][p] = asumR[j];
  __syncthreads();
  if (tid < K_CL) {
    float s = 0.0f;
    #pragma unroll
    for (int pp = 0; pp < T_PIX; ++pp) s += lgL[tid][pp];
    atomicAdd(&asum_ws[n * K_CL + tid], s);
  }
}

__global__ __launch_bounds__(256)
void netvlad_fin(const float* __restrict__ vlad_ws, const float* __restrict__ asum_ws,
                 const float* __restrict__ centroids, float* __restrict__ out) {
  __shared__ float rA[K_CL][4];
  __shared__ float rB[K_CL];
  const int n = blockIdx.x;
  const int tid = threadIdx.x;
  const int k = tid >> 2, q = tid & 3;
  const float* vb = vlad_ws + (size_t)n * K_CL * C_DIM;
  const float am = asum_ws[n * K_CL + k];

  float v[32];
  float ss = 0.0f;
  #pragma unroll
  for (int i = 0; i < 32; ++i) {
    const int c = q * 32 + i;
    const float val = vb[k * C_DIM + c] - am * centroids[k * C_DIM + c];
    v[i] = val;
    ss += val * val;
  }
  rA[k][q] = ss;
  __syncthreads();
  const float rss  = rA[k][0] + rA[k][1] + rA[k][2] + rA[k][3];
  const float rinv = 1.0f / fmaxf(sqrtf(rss), 1e-12f);
  if (q == 0) rB[k] = rss * rinv * rinv;
  __syncthreads();
  float gss = 0.0f;
  for (int kk = 0; kk < K_CL; ++kk) gss += rB[kk];
  const float ginv  = 1.0f / fmaxf(sqrtf(gss), 1e-12f);
  const float scale = rinv * ginv;
  #pragma unroll
  for (int i = 0; i < 32; ++i) {
    const int c = q * 32 + i;
    out[(size_t)n * (K_CL * C_DIM) + k * C_DIM + c] = v[i] * scale;
  }
}

extern "C" void kernel_launch(void* const* d_in, const int* in_sizes, int n_in,
                              void* d_out, int out_size, void* d_ws, size_t ws_size,
                              hipStream_t stream) {
  (void)in_sizes; (void)n_in; (void)out_size; (void)ws_size;
  const float* x         = (const float*)d_in[0];
  const float* conv_w    = (const float*)d_in[1];
  const float* centroids = (const float*)d_in[2];
  float* out = (float*)d_out;

  float* vlad_ws = (float*)d_ws;                         // [64][64][128] f32 = 2 MB
  float* asum_ws = vlad_ws + N_IMG * K_CL * C_DIM;       // [64][64] f32

  const int zcount = N_IMG * K_CL * C_DIM + N_IMG * K_CL;
  netvlad_zero<<<512, 256, 0, stream>>>((float*)d_ws, zcount);
  netvlad_main<<<N_IMG * CHUNKS, 256, 0, stream>>>(x, conv_w, vlad_ws, asum_ws);
  netvlad_fin <<<N_IMG, 256, 0, stream>>>(vlad_ws, asum_ws, centroids, out);
}
